// CausalSelfAttention_44848048505079
// MI455X (gfx1250) — compile-verified, matrix operand order checked by reading, not run on hardware
//
#include <hip/hip_runtime.h>

#define D_MODEL 1024
#define N_HEAD  16
#define D_HEAD  64
#define BATCH   2
#define SEQ     2048
#define M_TOTAL (BATCH * SEQ) /* 4096 */

#define BM 64
#define BN 128
#define BK 32

typedef __attribute__((ext_vector_type(16))) __bf16 v16bf;
typedef __attribute__((ext_vector_type(8)))  __bf16 v8bf;
typedef __attribute__((ext_vector_type(8)))  float  v8f;
typedef __attribute__((ext_vector_type(4)))  unsigned int v4u;
typedef __attribute__((ext_vector_type(8)))  int v8i;
typedef __attribute__((ext_vector_type(4)))  int v4i;

__device__ __forceinline__ v16bf combine16(v8bf lo, v8bf hi) {
  v16bf r;
#pragma unroll
  for (int i = 0; i < 8; ++i) { r[i] = lo[i]; r[i + 8] = hi[i]; }
  return r;
}

__device__ __forceinline__ v16bf load_frag(const __bf16* p0, const __bf16* p1) {
  return combine16(*(const v8bf*)p0, *(const v8bf*)p1);
}

__device__ __forceinline__ v8f wmma_bf16(v16bf a, v16bf b, v8f c) {
  return __builtin_amdgcn_wmma_f32_16x16x32_bf16(false, a, false, b, (short)0, c,
                                                 false, false);
}

// ---- Tensor Data Mover: 2D tile load (bf16) per ISA D# layout (groups 0/1) ----
// 6-arg builtin form: (v4u g0, v8i g1, v4i g2, v4i g3, v8i g4, i32 cpol)
// tensor_d0/d1, tile_d0/d1, stride0 are compile-time here so g1 folds to constants.
template <int TD0, int TD1, int TILE0, int TILE1>
__device__ __forceinline__ void tdm_load_2d(unsigned lds_off, const void* gptr) {
  unsigned long long ga = (unsigned long long)gptr;
  unsigned ga_lo = (unsigned)(ga & 0xffffffffu);
  unsigned ga_hi = (unsigned)((ga >> 32) & 0x01ffffffu) | (2u << 30); // type=2
  v4u g0;
  g0[0] = 1u;                                            // count=1, user D#
  g0[1] = __builtin_amdgcn_readfirstlane(lds_off);       // lds_addr (bytes)
  g0[2] = __builtin_amdgcn_readfirstlane(ga_lo);         // global_addr[31:0]
  g0[3] = __builtin_amdgcn_readfirstlane(ga_hi);         // global_addr[56:32]|type
  v8i g1;
  g1[0] = 1 << 16;                                       // data_size=1 (2 bytes)
  g1[1] = (TD0 & 0xffff) << 16;                          // tensor_dim0[15:0]
  g1[2] = ((TD0 >> 16) & 0xffff) | ((TD1 & 0xffff) << 16);
  g1[3] = ((TD1 >> 16) & 0xffff) | ((TILE0 & 0xffff) << 16);
  g1[4] = TILE1 & 0xffff;                                // tile_dim1 (tile_dim2=0)
  g1[5] = TD0;                                           // tensor_dim0_stride[31:0]
  g1[6] = 0;
  g1[7] = 0;
  v4i g2 = {0, 0, 0, 0};
  v4i g3 = {0, 0, 0, 0};
  v8i g4 = {0, 0, 0, 0, 0, 0, 0, 0};
  __builtin_amdgcn_tensor_load_to_lds(g0, g1, g2, g3, g4, 0);
}

// ---------------------------------------------------------------- prep kernels
__global__ void f32_to_bf16_kernel(const float* __restrict__ in,
                                   __bf16* __restrict__ out, int n) {
  int i = blockIdx.x * blockDim.x + threadIdx.x;
  if (i < n) out[i] = (__bf16)in[i];
}

__global__ void transpose_bf16_kernel(const float* __restrict__ in,
                                      __bf16* __restrict__ out, int rows, int cols) {
  int i = blockIdx.x * blockDim.x + threadIdx.x;
  int n = rows * cols;
  if (i < n) {
    int r = i / cols, c = i % cols;
    out[(size_t)c * rows + r] = (__bf16)in[(size_t)r * cols + c];
  }
}

// ---------------------------------------------------------------- tiled GEMM
// Workgroup: 256 threads (8 waves), block tile 64(M) x 128(N), K-step 32.
// Waves arranged 2(M) x 4(N); each wave computes a 2x2 grid of 16x16 WMMA tiles.
// A/B tiles are DMA'd into LDS by the Tensor Data Mover (double-buffered).
// MODE 0: QKV (N=3072, scatter q/k/vT bf16).  MODE 1: proj (N=1024, fp32 + bias).
template <int MODE>
__global__ __launch_bounds__(256, 1) void gemm_tiled_kernel(
    const __bf16* __restrict__ A, const __bf16* __restrict__ Bt,
    const float* __restrict__ bias,
    __bf16* __restrict__ qo, __bf16* __restrict__ ko, __bf16* __restrict__ vto,
    float* __restrict__ outf) {
  constexpr int KD = D_MODEL;                              // 1024
  constexpr int NCOLS = (MODE == 0) ? 3 * D_MODEL : D_MODEL;
  constexpr int MROWS = M_TOTAL;

  __shared__ __align__(16) __bf16 lds_a[2][BM * BK]; // 2 x 4KB
  __shared__ __align__(16) __bf16 lds_b[2][BN * BK]; // 2 x 8KB

  const int lane = threadIdx.x & 31;
  const int lm = lane & 15;
  const int hi = lane >> 4;
  const int wid = threadIdx.x >> 5; // 0..7
  const int wm = wid >> 2;          // 0..1
  const int wn = wid & 3;           // 0..3

  constexpr int ntb = NCOLS / BN;
  const int mb = blockIdx.x / ntb;
  const int nb = blockIdx.x % ntb;
  const int m0 = mb * BM, n0 = nb * BN;

  const int a_row0 = wm * 32;
  const int b_row0 = wn * 32;

  v8f acc00 = {}, acc01 = {}, acc10 = {}, acc11 = {};

  if (threadIdx.x < 32) {
    tdm_load_2d<KD, MROWS, BK, BM>((unsigned)(size_t)&lds_a[0][0],
                                   A + (size_t)m0 * KD);
    tdm_load_2d<KD, NCOLS, BK, BN>((unsigned)(size_t)&lds_b[0][0],
                                   Bt + (size_t)n0 * KD);
  }

#pragma unroll 2
  for (int k0 = 0; k0 < KD; k0 += BK) {
    const int cur = (k0 >> 5) & 1;
    if (threadIdx.x < 32) {
      if (k0 + BK < KD) {
        tdm_load_2d<KD, MROWS, BK, BM>((unsigned)(size_t)&lds_a[cur ^ 1][0],
                                       A + (size_t)m0 * KD + (k0 + BK));
        tdm_load_2d<KD, NCOLS, BK, BN>((unsigned)(size_t)&lds_b[cur ^ 1][0],
                                       Bt + (size_t)n0 * KD + (k0 + BK));
        __builtin_amdgcn_s_wait_tensorcnt(2); // current buffer's 2 DMAs done
      } else {
        __builtin_amdgcn_s_wait_tensorcnt(0);
      }
    }
    __syncthreads(); // LDS tiles visible to all 8 waves

    const __bf16* la = &lds_a[cur][0];
    const __bf16* lb = &lds_b[cur][0];
    const __bf16* la0 = la + (a_row0 + lm) * BK + hi * 8;
    const __bf16* la1 = la + (a_row0 + 16 + lm) * BK + hi * 8;
    const __bf16* lb0 = lb + (b_row0 + lm) * BK + hi * 8;
    const __bf16* lb1 = lb + (b_row0 + 16 + lm) * BK + hi * 8;
    v16bf a0 = load_frag(la0, la0 + 16);
    v16bf a1 = load_frag(la1, la1 + 16);
    v16bf b0 = load_frag(lb0, lb0 + 16);
    v16bf b1 = load_frag(lb1, lb1 + 16);
    acc00 = wmma_bf16(a0, b0, acc00);
    acc01 = wmma_bf16(a0, b1, acc01);
    acc10 = wmma_bf16(a1, b0, acc10);
    acc11 = wmma_bf16(a1, b1, acc11);

    __syncthreads(); // everyone done reading before next DMA overwrites
  }

  auto epilogue = [&](v8f acc, int ti, int tj) {
    const int col = n0 + b_row0 + tj * 16 + lm;
    const float bv = bias[col];
    if constexpr (MODE == 0) {
      const int which = col >> 10; // 0=q 1=k 2=v
      const int c1 = col & (D_MODEL - 1);
      const int h = c1 >> 6, dh = c1 & 63;
#pragma unroll
      for (int r = 0; r < 8; ++r) {
        const int row = m0 + a_row0 + ti * 16 + r + 8 * hi;
        const int b = row >> 11, t = row & (SEQ - 1);
        const int bh = b * N_HEAD + h;
        const __bf16 val = (__bf16)(acc[r] + bv);
        if (which == 0)
          qo[((size_t)bh * SEQ + t) * D_HEAD + dh] = val;
        else if (which == 1)
          ko[((size_t)bh * SEQ + t) * D_HEAD + dh] = val;
        else
          vto[((size_t)bh * D_HEAD + dh) * SEQ + t] = val;
      }
    } else {
#pragma unroll
      for (int r = 0; r < 8; ++r) {
        const int row = m0 + a_row0 + ti * 16 + r + 8 * hi;
        outf[(size_t)row * D_MODEL + col] = acc[r] + bv;
      }
    }
  };
  epilogue(acc00, 0, 0);
  epilogue(acc01, 0, 1);
  epilogue(acc10, 1, 0);
  epilogue(acc11, 1, 1);
}

// ---------------------------------------------------------------- attention
__global__ __launch_bounds__(128, 1) void attn_kernel(
    const __bf16* __restrict__ q, const __bf16* __restrict__ k,
    const __bf16* __restrict__ vt, __bf16* __restrict__ ao /* [B*T, D_MODEL] */) {
  __shared__ __align__(16) __bf16 pbuf[4 * 16 * 32];
  const int lane = threadIdx.x & 31;
  const int lm = lane & 15;
  const int hi = lane >> 4;
  const int wid = threadIdx.x >> 5;
  const int w = blockIdx.x * 4 + wid;
  const int bh = w / (SEQ / 16);
  const int i0 = (w % (SEQ / 16)) * 16;
  const int b = bh >> 4, h = bh & 15;

  const __bf16* Q = q + (size_t)bh * SEQ * D_HEAD;
  const __bf16* K = k + (size_t)bh * SEQ * D_HEAD;
  const __bf16* Vt = vt + (size_t)bh * D_HEAD * SEQ;

  const __bf16* qrow = Q + (size_t)(i0 + lm) * D_HEAD;
  const v16bf qa0 = load_frag(qrow + hi * 8, qrow + 16 + hi * 8);
  const v16bf qa1 = load_frag(qrow + 32 + hi * 8, qrow + 48 + hi * 8);

  v8f o0 = {}, o1 = {}, o2 = {}, o3 = {};
  float m[8], l[8];
#pragma unroll
  for (int r = 0; r < 8; ++r) { m[r] = -1e9f; l[r] = 0.0f; }

  __bf16* pb = pbuf + wid * 512;

  for (int j0 = 0; j0 < i0 + 16; j0 += 32) {
    // ---- scores, one n-half at a time (limits live K fragments)
    v8f s0 = {}, s1 = {};
    {
      const __bf16* krow0 = K + (size_t)(j0 + lm) * D_HEAD;
      v16bf kb00 = load_frag(krow0 + hi * 8, krow0 + 16 + hi * 8);
      v16bf kb01 = load_frag(krow0 + 32 + hi * 8, krow0 + 48 + hi * 8);
      s0 = wmma_bf16(qa0, kb00, s0);
      s0 = wmma_bf16(qa1, kb01, s0);
    }
    {
      const __bf16* krow1 = K + (size_t)(j0 + 16 + lm) * D_HEAD;
      v16bf kb10 = load_frag(krow1 + hi * 8, krow1 + 16 + hi * 8);
      v16bf kb11 = load_frag(krow1 + 32 + hi * 8, krow1 + 48 + hi * 8);
      s1 = wmma_bf16(qa0, kb10, s1);
      s1 = wmma_bf16(qa1, kb11, s1);
    }

    const int jc0 = j0 + lm, jc1 = j0 + 16 + lm;
#pragma unroll
    for (int r = 0; r < 8; ++r) {
      const int irow = i0 + r + 8 * hi;
      float v0 = s0[r] * 0.125f; // 1/sqrt(64)
      float v1 = s1[r] * 0.125f;
      if (jc0 > irow) v0 = -1e9f;
      if (jc1 > irow) v1 = -1e9f;
      s0[r] = v0; s1[r] = v1;
    }

    float alpha[8];
#pragma unroll
    for (int r = 0; r < 8; ++r) {
      float mb = fmaxf(s0[r], s1[r]);
      mb = fmaxf(mb, __shfl_xor(mb, 1));
      mb = fmaxf(mb, __shfl_xor(mb, 2));
      mb = fmaxf(mb, __shfl_xor(mb, 4));
      mb = fmaxf(mb, __shfl_xor(mb, 8));
      const float mn = fmaxf(m[r], mb);
      alpha[r] = __expf(m[r] - mn);
      m[r] = mn;
      const float p0 = __expf(s0[r] - mn);
      const float p1 = __expf(s1[r] - mn);
      s0[r] = p0; s1[r] = p1;
      float rs = p0 + p1;
      rs += __shfl_xor(rs, 1);
      rs += __shfl_xor(rs, 2);
      rs += __shfl_xor(rs, 4);
      rs += __shfl_xor(rs, 8);
      l[r] = l[r] * alpha[r] + rs;
    }
#pragma unroll
    for (int r = 0; r < 8; ++r) {
      o0[r] *= alpha[r]; o1[r] *= alpha[r];
      o2[r] *= alpha[r]; o3[r] *= alpha[r];
    }

    // ---- C-layout -> A-fragment relayout of P via per-wave LDS bounce
#pragma unroll
    for (int r = 0; r < 8; ++r) {
      const int row = r + 8 * hi;
      pb[row * 32 + lm]      = (__bf16)s0[r];
      pb[row * 32 + 16 + lm] = (__bf16)s1[r];
    }
    asm volatile("s_wait_dscnt 0" ::: "memory");
    v16bf pa = load_frag(pb + lm * 32 + hi * 8, pb + lm * 32 + 16 + hi * 8);

    // ---- O += P @ V (Vt rows are V columns => contiguous 16B loads)
    {
      const __bf16* vb0 = Vt + (size_t)(0 + lm) * SEQ + j0 + hi * 8;
      o0 = wmma_bf16(pa, load_frag(vb0, vb0 + 16), o0);
    }
    {
      const __bf16* vb1 = Vt + (size_t)(16 + lm) * SEQ + j0 + hi * 8;
      o1 = wmma_bf16(pa, load_frag(vb1, vb1 + 16), o1);
    }
    {
      const __bf16* vb2 = Vt + (size_t)(32 + lm) * SEQ + j0 + hi * 8;
      o2 = wmma_bf16(pa, load_frag(vb2, vb2 + 16), o2);
    }
    {
      const __bf16* vb3 = Vt + (size_t)(48 + lm) * SEQ + j0 + hi * 8;
      o3 = wmma_bf16(pa, load_frag(vb3, vb3 + 16), o3);
    }
  }

#pragma unroll
  for (int r = 0; r < 8; ++r) {
    const float rl = 1.0f / l[r];
    const int t = i0 + r + 8 * hi;
    const size_t rowoff = ((size_t)b * SEQ + t) * D_MODEL + h * D_HEAD + lm;
    ao[rowoff + 0]  = (__bf16)(o0[r] * rl);
    ao[rowoff + 16] = (__bf16)(o1[r] * rl);
    ao[rowoff + 32] = (__bf16)(o2[r] * rl);
    ao[rowoff + 48] = (__bf16)(o3[r] * rl);
  }
}

// ---------------------------------------------------------------- launch
extern "C" void kernel_launch(void* const* d_in, const int* in_sizes, int n_in,
                              void* d_out, int out_size, void* d_ws, size_t ws_size,
                              hipStream_t stream) {
  const float* x      = (const float*)d_in[0];
  /* d_in[1] = mask (causal, recomputed analytically) */
  const float* qkv_w  = (const float*)d_in[2];
  const float* qkv_b  = (const float*)d_in[3];
  const float* proj_w = (const float*)d_in[4];
  const float* proj_b = (const float*)d_in[5];
  float* out = (float*)d_out;

  char* ws = (char*)d_ws;
  __bf16* x_bf    = (__bf16*)(ws + (0ull  << 20)); // 8 MiB
  __bf16* qkv_wt  = (__bf16*)(ws + (8ull  << 20)); // 6 MiB  [3072,1024]
  __bf16* proj_wt = (__bf16*)(ws + (14ull << 20)); // 2 MiB  [1024,1024]
  __bf16* qb      = (__bf16*)(ws + (16ull << 20)); // 8 MiB  [B,H,T,Dh]
  __bf16* kb      = (__bf16*)(ws + (24ull << 20)); // 8 MiB  [B,H,T,Dh]
  __bf16* vtb     = (__bf16*)(ws + (32ull << 20)); // 8 MiB  [B,H,Dh,T]
  __bf16* attn_o  = (__bf16*)(ws + (40ull << 20)); // 8 MiB  [B*T, D_MODEL]

  const int n_x = M_TOTAL * D_MODEL;
  const int n_qkvw = D_MODEL * 3 * D_MODEL;
  const int n_projw = D_MODEL * D_MODEL;

  f32_to_bf16_kernel<<<(n_x + 255) / 256, 256, 0, stream>>>(x, x_bf, n_x);
  transpose_bf16_kernel<<<(n_qkvw + 255) / 256, 256, 0, stream>>>(
      qkv_w, qkv_wt, D_MODEL, 3 * D_MODEL);
  transpose_bf16_kernel<<<(n_projw + 255) / 256, 256, 0, stream>>>(
      proj_w, proj_wt, D_MODEL, D_MODEL);

  // QKV: (4096/64) x (3072/128) = 64 x 24 = 1536 blocks
  gemm_tiled_kernel<0><<<1536, 256, 0, stream>>>(x_bf, qkv_wt, qkv_b,
                                                 qb, kb, vtb, nullptr);

  // attention: 4096 q-tiles, 4 waves/block
  attn_kernel<<<4096 / 4, 128, 0, stream>>>(qb, kb, vtb, attn_o);

  // proj: (4096/64) x (1024/128) = 64 x 8 = 512 blocks
  gemm_tiled_kernel<1><<<512, 256, 0, stream>>>(attn_o, proj_wt, proj_b,
                                                nullptr, nullptr, nullptr, out);
}
